// KalmanPredictorSimpleMambaV1f_46591805227533
// MI455X (gfx1250) — matrix operand order repaired and checked by reading, not compiled, not run on hardware
//
#include <hip/hip_runtime.h>
#include <math.h>

#define B_     4
#define CIN    6
#define H_     64
#define W_     64
#define L_     (H_*W_)
#define DM     72
#define DI     144
#define KDIR   4
#define NST    8
#define RRANK  4
#define NLAY   2
#define GROUPS 12
#define CPG    (DM/GROUPS)

#define KP72   96
#define KP144  160
#define MP20   32
#define MP72   80

typedef __attribute__((ext_vector_type(16))) _Float16     v16h;
typedef __attribute__((ext_vector_type(8)))  _Float16     h8;
typedef __attribute__((ext_vector_type(8)))  float        v8f;
typedef __attribute__((ext_vector_type(4)))  unsigned int u32x4;
typedef __attribute__((ext_vector_type(8)))  int          i32x8;
typedef __attribute__((ext_vector_type(4)))  int          i32x4;

#ifndef __has_builtin
#define __has_builtin(x) 0
#endif
#if __has_builtin(__builtin_amdgcn_tensor_load_to_lds)
#define USE_TDM 1
#else
#define USE_TDM 0
#endif

__device__ __forceinline__ float lrelu_f(float x, float s){ return x >= 0.f ? x : s*x; }
__device__ __forceinline__ float silu_f(float x){ return x / (1.f + __expf(-x)); }
__device__ __forceinline__ float sigmoid_f(float x){ return 1.f / (1.f + __expf(-x)); }
__device__ __forceinline__ float softplus_f(float x){ return x > 20.f ? x : log1pf(__expf(x)); }

__device__ __forceinline__ void wait_tensorcnt0()
{
#if __has_builtin(__builtin_amdgcn_s_wait_tensorcnt)
    __builtin_amdgcn_s_wait_tensorcnt(0);
#else
    asm volatile("s_wait_tensorcnt 0x0" ::: "memory");
#endif
}

// ---------------------------------------------------------------------------
// WMMA GEMM on pre-padded f16 operands.
// Block = 128 threads (4 waves): 4 M-tiles x 1 N-tile, sharing one 32x16 f16
// B tile staged in LDS by the Tensor Data Mover (TENSORcnt) each K-step.
// ---------------------------------------------------------------------------
__global__ void gemm_wmma_kernel(const _Float16* __restrict__ A, const _Float16* __restrict__ Bm,
                                 float* __restrict__ C, _Float16* __restrict__ C16,
                                 const float* __restrict__ bias,
                                 int M, int Mpad, int Kpad, int N,
                                 long long sBz, long long sCz, long long sC16z,
                                 int act, float slope)
{
    __shared__ __align__(16) _Float16 sB[32*16];   // row-major K x 16 tile

    const int z = blockIdx.z;
    Bm += (long long)z * sBz;
    C  += (long long)z * sCz;
    if (C16) C16 += (long long)z * sC16z;

    const int numMt = Mpad >> 4;
    const int wave  = threadIdx.x >> 5;
    const int lane  = threadIdx.x & 31;
    const int nt = blockIdx.x;
    const int mt = blockIdx.y*4 + wave;
    const bool waveActive = (mt < numMt);
    const int m0 = mt << 4;
    const int n0 = nt << 4;
    const int grp  = lane >> 4;
    const int mrow = m0 + (lane & 15);
    const int ncol = n0 + (lane & 15);

    const _Float16* arow = A + (long long)(waveActive ? mrow : 0)*Kpad;
    const unsigned ldsa = (unsigned)(unsigned long long)(void*)sB;

    v8f acc = {};
    for (int k0 = 0; k0 < Kpad; k0 += 32) {
        // ---- stage B tile rows [k0,k0+32) x cols [n0,n0+16) into LDS ----
#if USE_TDM
        if (wave == 0) {
            const unsigned long long g =
                (unsigned long long)(const void*)(Bm + (long long)k0*N + n0);
            u32x4 g0; i32x8 g1; i32x4 g2; i32x4 g3;
            g0[0] = 1u;                                  // count=1, user descriptor
            g0[1] = ldsa;                                // lds_addr (bytes)
            g0[2] = (unsigned)g;                         // global_addr[31:0]
            g0[3] = (unsigned)((g >> 32) & 0x1FFFFFFull) // global_addr[56:32]
                  | (2u << 30);                          // type=2 ("image")
            g1[0] = (1 << 16);                           // data_size=1 -> 2 bytes
            g1[1] = (N & 0xFFFF) << 16;                  // tensor_dim0[15:0]
            g1[2] = ((N >> 16) & 0xFFFF)                 // tensor_dim0[31:16]
                  | (32 << 16);                          // tensor_dim1[15:0] = 32
            g1[3] = (16 << 16);                          // tile_dim0 = 16
            g1[4] = 32;                                  // tile_dim1 = 32
            g1[5] = N;                                   // tensor_dim0_stride[31:0]
            g1[6] = 0;                                   // stride hi / dim1_stride lo
            g1[7] = 0;
            g2[0]=g2[1]=g2[2]=g2[3]=0;
            g3[0]=g3[1]=g3[2]=g3[3]=0;
#if defined(__clang_major__) && (__clang_major__ >= 23)
            i32x8 g4; g4[0]=g4[1]=g4[2]=g4[3]=g4[4]=g4[5]=g4[6]=g4[7]=0;
            __builtin_amdgcn_tensor_load_to_lds(g0, g1, g2, g3, g4, 0);
#else
            __builtin_amdgcn_tensor_load_to_lds(g0, g1, g2, g3, 0);
#endif
            wait_tensorcnt0();
        }
#else
        // fallback: cooperative copy, 128 threads x 8 bytes
        {
            const int kk = threadIdx.x >> 2;          // 0..31
            const int cc = (threadIdx.x & 3) * 4;     // 0,4,8,12
            *(unsigned long long*)(sB + kk*16 + cc) =
                *(const unsigned long long*)(Bm + (long long)(k0 + kk)*N + n0 + cc);
        }
#endif
        __syncthreads();

        if (waveActive) {
            // ---- A fragment: two aligned 16B vector loads (zero-padded K) ----
            h8 alo = *(const h8*)(arow + k0 + 8*grp);
            h8 ahi = *(const h8*)(arow + k0 + 16 + 8*grp);
            v16h a;
            #pragma unroll
            for (int i = 0; i < 8; ++i) { a[i] = alo[i]; a[8+i] = ahi[i]; }

            // ---- B fragment from LDS: column = lane&15, K = 8*grp + i (i<8), 16 + 8*grp + (i-8) (i>=8) ----
            v16h b;
            #pragma unroll
            for (int i = 0; i < 16; ++i) b[i] = sB[((i < 8) ? (8*grp + i) : (16 + 8*grp + (i - 8)))*16 + (lane & 15)];

            if (k0 + 32 < Kpad)
                __builtin_prefetch(arow + k0 + 32, 0, 1);

            acc = __builtin_amdgcn_wmma_f32_16x16x32_f16(
                      false, a, false, b, (short)0, acc, false, false);
        }
        __syncthreads();
    }

    if (waveActive) {
        #pragma unroll
        for (int r = 0; r < 8; ++r) {
            const int row = m0 + r + 8*grp;
            if (row < M) {
                float v = acc[r];
                if (bias) v += bias[row];
                if (act == 1) v = lrelu_f(v, slope);
                C[(long long)row*N + ncol] = v;
                if (C16) C16[(long long)row*N + ncol] = (_Float16)v;
            }
        }
    }
}

__global__ void cvt_pad_k(const float* __restrict__ src, _Float16* __restrict__ dst,
                          int M, int K, int Mpad, int Kpad)
{
    int idx = blockIdx.x*blockDim.x + threadIdx.x;
    const int total = Mpad*Kpad;
    if (idx >= total) return;
    const int m = idx / Kpad, k = idx % Kpad;
    dst[idx] = (m < M && k < K) ? (_Float16)src[m*K + k] : (_Float16)0.f;
}

__global__ void zero_u32_k(unsigned int* __restrict__ p, long long n)
{
    long long i = (long long)blockIdx.x*blockDim.x + threadIdx.x;
    if (i < n) p[i] = 0u;
}

__global__ void conv_in_k(const float* __restrict__ img, const float* __restrict__ w,
                          const float* __restrict__ bias, float* __restrict__ out)
{
    int idx = blockIdx.x*blockDim.x + threadIdx.x;
    const int total = B_*DM*H_*W_;
    if (idx >= total) return;
    const int x = idx % W_, y = (idx/W_) % H_, o = (idx/(W_*H_)) % DM, b = idx/(W_*H_*DM);
    float s = bias[o];
    for (int c = 0; c < CIN; ++c) {
        const float* ip = img + ((long long)(b*CIN + c))*H_*W_;
        const float* wp = w + (long long)(o*CIN + c)*9;
        #pragma unroll
        for (int ky = 0; ky < 3; ++ky) {
            const int yy = y + ky - 1; if ((unsigned)yy >= (unsigned)H_) continue;
            #pragma unroll
            for (int kx = 0; kx < 3; ++kx) {
                const int xx = x + kx - 1; if ((unsigned)xx >= (unsigned)W_) continue;
                s += wp[ky*3+kx] * ip[yy*W_ + xx];
            }
        }
    }
    out[idx] = lrelu_f(s, 0.01f);
}

__global__ void conv_mid_k(const float* __restrict__ h1, const float* __restrict__ img,
                           const float* __restrict__ w2, const float* __restrict__ b2,
                           const float* __restrict__ ws, const float* __restrict__ bs,
                           float* __restrict__ out, _Float16* __restrict__ out16)
{
    int idx = blockIdx.x*blockDim.x + threadIdx.x;
    const int total = B_*DM*H_*W_;
    if (idx >= total) return;
    const int x = idx % W_, y = (idx/W_) % H_, o = (idx/(W_*H_)) % DM, b = idx/(W_*H_*DM);
    float s = b2[o] + bs[o];
    for (int c = 0; c < DM; ++c) {
        const float* ip = h1 + ((long long)(b*DM + c))*H_*W_;
        const float* wp = w2 + (long long)(o*DM + c)*9;
        #pragma unroll
        for (int ky = 0; ky < 3; ++ky) {
            const int yy = y + ky - 1; if ((unsigned)yy >= (unsigned)H_) continue;
            #pragma unroll
            for (int kx = 0; kx < 3; ++kx) {
                const int xx = x + kx - 1; if ((unsigned)xx >= (unsigned)W_) continue;
                s += wp[ky*3+kx] * ip[yy*W_ + xx];
            }
        }
    }
    for (int c = 0; c < CIN; ++c)
        s += ws[o*CIN + c] * img[((long long)(b*CIN + c))*H_*W_ + y*W_ + x];
    s = lrelu_f(s, 0.01f);
    out[idx] = s;
    out16[((long long)b*KP72 + o)*L_ + y*W_ + x] = (_Float16)s;
}

__global__ void conv_out_k(const float* __restrict__ xin, const float* __restrict__ w,
                           const float* __restrict__ bias, float* __restrict__ out)
{
    int idx = blockIdx.x*blockDim.x + threadIdx.x;
    const int total = B_*CIN*H_*W_;
    if (idx >= total) return;
    const int x = idx % W_, y = (idx/W_) % H_, o = (idx/(W_*H_)) % CIN, b = idx/(W_*H_*CIN);
    float s = bias[o];
    for (int c = 0; c < DM; ++c) {
        const float* ip = xin + ((long long)(b*DM + c))*H_*W_;
        const float* wp = w + (long long)(o*DM + c)*9;
        #pragma unroll
        for (int ky = 0; ky < 3; ++ky) {
            const int yy = y + ky - 1; if ((unsigned)yy >= (unsigned)H_) continue;
            #pragma unroll
            for (int kx = 0; kx < 3; ++kx) {
                const int xx = x + kx - 1; if ((unsigned)xx >= (unsigned)W_) continue;
                s += wp[ky*3+kx] * ip[yy*W_ + xx];
            }
        }
    }
    out[idx] = sigmoid_f(s);
}

__global__ void dwconv_silu_k(const float* __restrict__ xz, const float* __restrict__ w,
                              const float* __restrict__ bias,
                              float* __restrict__ x0, float* __restrict__ xt,
                              _Float16* __restrict__ x016, _Float16* __restrict__ xt16)
{
    int idx = blockIdx.x*blockDim.x + threadIdx.x;
    const int total = B_*DI*H_*W_;
    if (idx >= total) return;
    const int x = idx % W_, y = (idx/W_) % H_, d = (idx/(W_*H_)) % DI, b = idx/(W_*H_*DI);
    const float* ip = xz + ((long long)b*(2*DI) + d)*L_;
    const float* wp = w + d*9;
    float s = bias[d];
    #pragma unroll
    for (int ky = 0; ky < 3; ++ky) {
        const int yy = y + ky - 1; if ((unsigned)yy >= (unsigned)H_) continue;
        #pragma unroll
        for (int kx = 0; kx < 3; ++kx) {
            const int xx = x + kx - 1; if ((unsigned)xx >= (unsigned)W_) continue;
            s += wp[ky*3+kx] * ip[yy*W_ + xx];
        }
    }
    s = silu_f(s);
    const int ln = y*W_ + x;
    const int lt = x*H_ + y;
    x0 [((long long)b*DI    + d)*L_ + ln] = s;
    xt [((long long)b*DI    + d)*L_ + lt] = s;
    x016[((long long)b*KP144 + d)*L_ + ln] = (_Float16)s;
    xt16[((long long)b*KP144 + d)*L_ + lt] = (_Float16)s;
}

#define DCH 4
#define CPB 36
__global__ void scan_k(const float* __restrict__ x0, const float* __restrict__ xt,
                       const float* __restrict__ xdbl,
                       const float* __restrict__ dtp_w, const float* __restrict__ dtp_b,
                       const float* __restrict__ A_logs, const float* __restrict__ Ds,
                       float* __restrict__ y_dir)
{
    const int z = blockIdx.x;
    const int chunk = z % DCH;
    const int bk = z / DCH;
    const int k  = bk % KDIR;
    const int b  = bk / KDIR;
    const int dl = chunk*CPB + threadIdx.x;
    const bool active = (threadIdx.x < CPB) && (dl < DI);
    const int d = active ? dl : 0;

    const float* u_base = ((k & 1) ? xt : x0) + ((long long)b*DI + d)*L_;
    const float* G = xdbl + ((long long)(b*KDIR + k)*20)*L_;
    float* yp = y_dir + ((long long)(b*KDIR + k)*DI + d)*L_;

    float wdt[RRANK], A_row[NST], h[NST];
    const float bdt = dtp_b[k*DI + d];
    const float Dk  = Ds[k*DI + d];
    #pragma unroll
    for (int r = 0; r < RRANK; ++r) wdt[r] = dtp_w[(k*DI + d)*RRANK + r];
    #pragma unroll
    for (int n = 0; n < NST; ++n) { A_row[n] = -expf(A_logs[(k*DI + d)*NST + n]); h[n] = 0.f; }

    __shared__ float s[20];
    const bool fwd = (k < 2);
    for (int t = 0; t < L_; ++t) {
        const int l = fwd ? t : (L_ - 1 - t);
        if (threadIdx.x < 20) s[threadIdx.x] = G[threadIdx.x*L_ + l];
        __syncthreads();
        if (active) {
            const float dt = softplus_f(wdt[0]*s[0] + wdt[1]*s[1] + wdt[2]*s[2] + wdt[3]*s[3] + bdt);
            const float u  = u_base[l];
            const float du = dt*u;
            float yv = 0.f;
            #pragma unroll
            for (int n = 0; n < NST; ++n) {
                h[n] = h[n]*__expf(dt*A_row[n]) + du*s[4+n];
                yv  += h[n]*s[12+n];
            }
            yp[l] = yv + u*Dk;
        }
        __syncthreads();
    }
}

__global__ void combine_ln_k(const float* __restrict__ y_dir, const float* __restrict__ xz,
                             const float* __restrict__ ong, const float* __restrict__ onb,
                             _Float16* __restrict__ y_act16)
{
    const int pos = blockIdx.x;
    const int b = pos / L_;
    const int l = pos % L_;
    const int hh = l >> 6, ww = l & 63;
    const int lt = (ww << 6) | hh;
    const int tid = threadIdx.x;

    __shared__ float sv[DI];
    __shared__ float r1[256], r2[256];

    float v = 0.f;
    if (tid < DI) {
        const long long base = (long long)b*KDIR*DI*L_;
        v = y_dir[base + ((long long)(0*DI + tid))*L_ + l]
          + y_dir[base + ((long long)(2*DI + tid))*L_ + l]
          + y_dir[base + ((long long)(1*DI + tid))*L_ + lt]
          + y_dir[base + ((long long)(3*DI + tid))*L_ + lt];
        sv[tid] = v;
    }
    r1[tid] = (tid < DI) ? v   : 0.f;
    r2[tid] = (tid < DI) ? v*v : 0.f;
    __syncthreads();
    for (int s = 128; s > 0; s >>= 1) {
        if (tid < s) { r1[tid] += r1[tid+s]; r2[tid] += r2[tid+s]; }
        __syncthreads();
    }
    const float mu   = r1[0] / (float)DI;
    const float var  = r2[0] / (float)DI - mu*mu;
    const float rstd = rsqrtf(var + 1e-5f);
    if (tid < DI) {
        const float zg = xz[((long long)b*(2*DI) + DI + tid)*L_ + l];
        float ya = (sv[tid] - mu)*rstd*ong[tid] + onb[tid];
        ya *= silu_f(zg);
        y_act16[((long long)b*KP144 + tid)*L_ + l] = (_Float16)ya;
    }
}

__global__ void groupnorm_k(const float* __restrict__ xin, const float* __restrict__ g,
                            const float* __restrict__ bta, _Float16* __restrict__ out16)
{
    const int b  = blockIdx.x / GROUPS;
    const int gi = blockIdx.x % GROUPS;
    const float* base    = xin   + ((long long)b*DM   + gi*CPG)*L_;
    _Float16*    obase16 = out16 + ((long long)b*KP72 + gi*CPG)*L_;
    const int count = CPG*L_;
    float sum = 0.f, sq = 0.f;
    for (int i = threadIdx.x; i < count; i += blockDim.x) {
        const float v = base[i]; sum += v; sq += v*v;
    }
    __shared__ float r1[256], r2[256];
    r1[threadIdx.x] = sum; r2[threadIdx.x] = sq;
    __syncthreads();
    for (int s = 128; s > 0; s >>= 1) {
        if (threadIdx.x < s) { r1[threadIdx.x] += r1[threadIdx.x+s]; r2[threadIdx.x] += r2[threadIdx.x+s]; }
        __syncthreads();
    }
    const float mu   = r1[0] / (float)count;
    const float var  = r2[0] / (float)count - mu*mu;
    const float rstd = rsqrtf(var + 1e-5f);
    for (int i = threadIdx.x; i < count; i += blockDim.x) {
        const int c = gi*CPG + i / L_;
        obase16[i] = (_Float16)((base[i] - mu)*rstd*g[c] + bta[c]);
    }
}

static inline void launch_gemm(const _Float16* A, const _Float16* Bm, float* C, _Float16* C16,
                               const float* bias, int M, int Mpad, int Kpad, int N,
                               long long sBz, long long sCz, long long sC16z,
                               int batches, int act, float slope, hipStream_t stream)
{
    const int numMt = Mpad/16, numNt = N/16;
    dim3 grid(numNt, (numMt + 3)/4, batches);
    gemm_wmma_kernel<<<grid, 128, 0, stream>>>(A, Bm, C, C16, bias,
                                               M, Mpad, Kpad, N, sBz, sCz, sC16z, act, slope);
}

extern "C" void kernel_launch(void* const* d_in, const int* in_sizes, int n_in,
                              void* d_out, int out_size, void* d_ws, size_t ws_size,
                              hipStream_t stream)
{
    (void)in_sizes; (void)n_in; (void)out_size; (void)ws_size;

    const float* image   = (const float*)d_in[0];
    const float* cb_w1   = (const float*)d_in[1];
    const float* cb_b1   = (const float*)d_in[2];
    const float* cb_w2   = (const float*)d_in[3];
    const float* cb_b2   = (const float*)d_in[4];
    const float* cb_ws   = (const float*)d_in[5];
    const float* cb_bs   = (const float*)d_in[6];
    const float* in_w    = (const float*)d_in[7];
    const float* dw_w    = (const float*)d_in[8];
    const float* dw_b    = (const float*)d_in[9];
    const float* xp_w    = (const float*)d_in[10];
    const float* dtp_w   = (const float*)d_in[11];
    const float* dtp_b   = (const float*)d_in[12];
    const float* A_logs  = (const float*)d_in[13];
    const float* Ds      = (const float*)d_in[14];
    const float* ong     = (const float*)d_in[15];
    const float* onb     = (const float*)d_in[16];
    const float* op_w    = (const float*)d_in[17];
    const float* gn_g    = (const float*)d_in[18];
    const float* gn_b    = (const float*)d_in[19];
    const float* l1_w    = (const float*)d_in[20];
    const float* l1_b    = (const float*)d_in[21];
    const float* l2_w    = (const float*)d_in[22];
    const float* l2_b    = (const float*)d_in[23];
    const float* shrink_w= (const float*)d_in[24];
    const float* shrink_b= (const float*)d_in[25];

    char* base = (char*)d_ws;
    size_t off = 0;
    auto allocB = [&](size_t bytes) -> char* {
        char* p = base + off;
        off += (bytes + 255) & ~(size_t)255;
        return p;
    };
    float* xbuf = (float*)allocB(sizeof(float)*(size_t)B_*DM*L_);
    float* bufA = (float*)allocB(sizeof(float)*(size_t)B_*DM*L_);
    float* xz   = (float*)allocB(sizeof(float)*(size_t)B_*2*DI*L_);
    float* x0   = (float*)allocB(sizeof(float)*(size_t)B_*DI*L_);
    float* xt   = (float*)allocB(sizeof(float)*(size_t)B_*DI*L_);
    float* xdbl = (float*)allocB(sizeof(float)*(size_t)B_*KDIR*20*L_);
    float* ydir = (float*)allocB(sizeof(float)*(size_t)B_*KDIR*DI*L_);

    char* f16_start = base + off;
    _Float16* xbuf16 = (_Float16*)allocB(sizeof(_Float16)*(size_t)B_*KP72 *L_);
    _Float16* x016   = (_Float16*)allocB(sizeof(_Float16)*(size_t)B_*KP144*L_);
    _Float16* xt16   = (_Float16*)allocB(sizeof(_Float16)*(size_t)B_*KP144*L_);
    _Float16* yact16 = (_Float16*)allocB(sizeof(_Float16)*(size_t)B_*KP144*L_);
    _Float16* bufA16 = (_Float16*)allocB(sizeof(_Float16)*(size_t)B_*KP72 *L_);
    _Float16* bufB16 = (_Float16*)allocB(sizeof(_Float16)*(size_t)B_*KP72 *L_);
    char* f16_end = base + off;

    _Float16* in_w16[NLAY]; _Float16* xp_w16[NLAY];
    _Float16* op_w16[NLAY]; _Float16* l1_w16[NLAY]; _Float16* l2_w16[NLAY];
    for (int i = 0; i < NLAY; ++i) {
        in_w16[i] = (_Float16*)allocB(sizeof(_Float16)*(size_t)288*KP72);
        xp_w16[i] = (_Float16*)allocB(sizeof(_Float16)*(size_t)KDIR*MP20*KP144);
        op_w16[i] = (_Float16*)allocB(sizeof(_Float16)*(size_t)MP72*KP144);
        l1_w16[i] = (_Float16*)allocB(sizeof(_Float16)*(size_t)MP72*KP72);
        l2_w16[i] = (_Float16*)allocB(sizeof(_Float16)*(size_t)MP72*KP72);
    }

    {
        long long n = (long long)(f16_end - f16_start) / 4;
        zero_u32_k<<<(unsigned)((n + 255)/256), 256, 0, stream>>>((unsigned int*)f16_start, n);
    }

    for (int i = 0; i < NLAY; ++i) {
        const float* in_w_i = in_w + (size_t)i*2*DI*DM;
        const float* xp_w_i = xp_w + (size_t)i*KDIR*20*DI;
        const float* op_w_i = op_w + (size_t)i*DM*DI;
        const float* l1_w_i = l1_w + (size_t)i*DM*DM;
        const float* l2_w_i = l2_w + (size_t)i*DM*DM;
        cvt_pad_k<<<(288*KP72 + 255)/256, 256, 0, stream>>>(in_w_i, in_w16[i], 2*DI, DM, 288, KP72);
        for (int k = 0; k < KDIR; ++k)
            cvt_pad_k<<<(MP20*KP144 + 255)/256, 256, 0, stream>>>(
                xp_w_i + (size_t)k*20*DI, xp_w16[i] + (size_t)k*MP20*KP144, 20, DI, MP20, KP144);
        cvt_pad_k<<<(MP72*KP144 + 255)/256, 256, 0, stream>>>(op_w_i, op_w16[i], DM, DI, MP72, KP144);
        cvt_pad_k<<<(MP72*KP72 + 255)/256, 256, 0, stream>>>(l1_w_i, l1_w16[i], DM, DM, MP72, KP72);
        cvt_pad_k<<<(MP72*KP72 + 255)/256, 256, 0, stream>>>(l2_w_i, l2_w16[i], DM, DM, MP72, KP72);
    }

    {
        const int total = B_*DM*H_*W_;
        conv_in_k<<<(total+255)/256, 256, 0, stream>>>(image, cb_w1, cb_b1, bufA);
        conv_mid_k<<<(total+255)/256, 256, 0, stream>>>(bufA, image, cb_w2, cb_b2, cb_ws, cb_bs,
                                                        xbuf, xbuf16);
    }

    for (int i = 0; i < NLAY; ++i) {
        const float* dw_w_i  = dw_w   + (size_t)i*DI*9;
        const float* dw_b_i  = dw_b   + (size_t)i*DI;
        const float* dtp_w_i = dtp_w  + (size_t)i*KDIR*DI*RRANK;
        const float* dtp_b_i = dtp_b  + (size_t)i*KDIR*DI;
        const float* Al_i    = A_logs + (size_t)i*KDIR*DI*NST;
        const float* Ds_i    = Ds     + (size_t)i*KDIR*DI;
        const float* ong_i   = ong    + (size_t)i*DI;
        const float* onb_i   = onb    + (size_t)i*DI;
        const float* gn_g_i  = gn_g   + (size_t)i*DM;
        const float* gn_b_i  = gn_b   + (size_t)i*DM;
        const float* l1_b_i  = l1_b   + (size_t)i*DM;
        const float* l2_b_i  = l2_b   + (size_t)i*DM;

        launch_gemm(in_w16[i], xbuf16, xz, nullptr, nullptr,
                    2*DI, 288, KP72, L_,
                    (long long)KP72*L_, (long long)2*DI*L_, 0, B_, 0, 0.f, stream);

        {
            const int total = B_*DI*H_*W_;
            dwconv_silu_k<<<(total+255)/256, 256, 0, stream>>>(xz, dw_w_i, dw_b_i,
                                                               x0, xt, x016, xt16);
        }

        for (int k = 0; k < KDIR; ++k) {
            launch_gemm(xp_w16[i] + (size_t)k*MP20*KP144, (k & 1) ? xt16 : x016,
                        xdbl + (size_t)k*20*L_, nullptr, nullptr,
                        20, MP20, KP144, L_,
                        (long long)KP144*L_, (long long)KDIR*20*L_, 0, B_, 0, 0.f, stream);
        }

        scan_k<<<B_*KDIR*DCH, 64, 0, stream>>>(x0, xt, xdbl, dtp_w_i, dtp_b_i, Al_i, Ds_i, ydir);

        combine_ln_k<<<B_*L_, 256, 0, stream>>>(ydir, xz, ong_i, onb_i, yact16);

        launch_gemm(op_w16[i], yact16, bufA, nullptr, nullptr,
                    DM, MP72, KP144, L_,
                    (long long)KP144*L_, (long long)DM*L_, 0, B_, 0, 0.f, stream);

        groupnorm_k<<<B_*GROUPS, 256, 0, stream>>>(bufA, gn_g_i, gn_b_i, bufB16);

        launch_gemm(l1_w16[i], bufB16, bufA, bufA16, l1_b_i,
                    DM, MP72, KP72, L_,
                    (long long)KP72*L_, (long long)DM*L_, (long long)KP72*L_,
                    B_, 1, 0.04f, stream);

        launch_gemm(l2_w16[i], bufA16, xbuf, xbuf16, l2_b_i,
                    DM, MP72, KP72, L_,
                    (long long)KP72*L_, (long long)DM*L_, (long long)KP72*L_,
                    B_, 0, 0.f, stream);
    }

    {
        const int total = B_*CIN*H_*W_;
        conv_out_k<<<(total+255)/256, 256, 0, stream>>>(xbuf, shrink_w, shrink_b, (float*)d_out);
    }
}
